// EncoderLayer_26242250178869
// MI455X (gfx1250) — compile-verified, matrix operand order checked by reading, not run on hardware
//
#include <hip/hip_runtime.h>

// ---------------------------------------------------------------------------
// Types for CDNA5 WMMA (wave32): v_wmma_f32_16x16x32_bf16
// ---------------------------------------------------------------------------
typedef __attribute__((ext_vector_type(16))) __bf16        v16bf;
typedef __attribute__((ext_vector_type(8)))  float         v8f;
typedef __attribute__((ext_vector_type(4)))  int           v4i_t;
typedef __attribute__((ext_vector_type(4)))  unsigned int  v4u_t;

union FragBF16 {
    v16bf v;
    unsigned int u[8];
};

__device__ __forceinline__ unsigned short f2bf(float f) {
    unsigned int u = __float_as_uint(f);
    unsigned int r = u + 0x7FFFu + ((u >> 16) & 1u);   // round-to-nearest-even
    return (unsigned short)(r >> 16);
}

// Global-address-space vector load (forces global_load_b128, not flat_load)
typedef __attribute__((address_space(1))) const v4u_t* g_v4u_cp;
__device__ __forceinline__ v4u_t gload4(const void* p) {
    return *(g_v4u_cp)(size_t)p;
}

// ---------------------------------------------------------------------------
// gfx1250 async global->LDS staging (ASYNCcnt path). Builtin when available,
// else inline asm (portable across ROCm / amdgpu-toolchain per bridge doc).
// Builtin signature (from hipcc diagnostic): (as1 int4*, as3 int4*, imm, imm).
// ---------------------------------------------------------------------------
#if defined(__has_builtin)
#if __has_builtin(__builtin_amdgcn_global_load_async_to_lds_b128)
#define HAVE_ASYNC_BUILTIN 1
#endif
#if __has_builtin(__builtin_amdgcn_s_wait_asynccnt)
#define HAVE_WAIT_ASYNC_BUILTIN 1
#endif
#endif

__device__ __forceinline__ void async_ld_b128(const void* g, void* lds) {
#if defined(HAVE_ASYNC_BUILTIN)
    typedef __attribute__((address_space(1))) v4i_t* gp_t;
    typedef __attribute__((address_space(3))) v4i_t* lp_t;
    __builtin_amdgcn_global_load_async_to_lds_b128(
        (gp_t)(size_t)g, (lp_t)(unsigned int)(size_t)lds, 0, 0);
#else
    unsigned int l = (unsigned int)(size_t)lds;                 // LDS byte offset
    unsigned long long a = (unsigned long long)(size_t)g;
    asm volatile("global_load_async_to_lds_b128 %0, %1, off"
                 :: "v"(l), "v"(a) : "memory");
#endif
}

__device__ __forceinline__ void wait_async0() {
#if defined(HAVE_WAIT_ASYNC_BUILTIN)
    __builtin_amdgcn_s_wait_asynccnt(0);
#else
    asm volatile("s_wait_asynccnt 0x0" ::: "memory");
#endif
}

// ---------------------------------------------------------------------------
// fp32 -> bf16 elementwise conversion
// ---------------------------------------------------------------------------
__global__ void cvt_f32_bf16(const float* __restrict__ in,
                             unsigned short* __restrict__ out, long long n) {
    long long i = (long long)blockIdx.x * blockDim.x + threadIdx.x;
    long long stride = (long long)gridDim.x * blockDim.x;
    for (; i < n; i += stride) out[i] = f2bf(in[i]);
}

// ---------------------------------------------------------------------------
// Generic bf16 WMMA GEMM:  C = act(alpha * A[M,K] @ B[K,N] + bias)
//   TRANSB: B given as [N,K] row-major (used for Q @ K^T)
//   flags bit0: output bf16 (else fp32), bit1: ELU activation
//   cstride: output column stride (8 for interleaved head writes)
// Block tile 128x128x32, 8 wave32 waves, each wave 2x4 tiles of 16x16.
// Double-buffered LDS; A (and B when TRANSB) staged via async-to-LDS.
// ---------------------------------------------------------------------------
#define BM 128
#define BN 128
#define BK 32
#define LDSA 40   // bf16 elements per LDS row (padded vs 32 to skew banks)
#define LDSB 40

template <bool TRANSB>
__global__ __launch_bounds__(256)
void gemm_bf16_wmma(const unsigned short* __restrict__ A, long long aBatch, int lda,
                    const unsigned short* __restrict__ B, long long bBatch, int ldb,
                    void* __restrict__ C, long long cBatch, int ldc, int cstride,
                    int M, int N, int K, float alpha,
                    const float* __restrict__ bias, int flags) {
    __shared__ unsigned short sA[2][BM * LDSA];     // [m][k]
    __shared__ unsigned short sB[2][BN * LDSB];     // [n][k] (transposed stage)

    const int tid   = threadIdx.x;
    const int lane  = tid & 31;
    const int wave  = tid >> 5;
    const int wx    = wave & 1;      // N half (64 cols)
    const int wy    = wave >> 1;     // M quarter (32 rows)
    const int l16   = lane & 15;
    const int lhalf = lane >> 4;     // K-half select per ISA 16-bit layout
    const int m0    = blockIdx.y * BM;
    const int n0    = blockIdx.x * BN;

    A += blockIdx.z * aBatch;
    B += blockIdx.z * bBatch;

    // staging coordinates
    const int arow = tid >> 1, aseg = tid & 1;      // A / TRANSB-B: row + 16-elem seg
    const int bkc  = tid & 31, bnb = (tid >> 5) * 16;  // non-trans B: k col + n seg

    v8f acc[2][4];
#pragma unroll
    for (int mi = 0; mi < 2; mi++)
#pragma unroll
        for (int ni = 0; ni < 4; ni++)
#pragma unroll
            for (int j = 0; j < 8; j++) acc[mi][ni][j] = 0.0f;

    const v4u_t zero4 = {0u, 0u, 0u, 0u};

    auto stageA = [&](int buf, int k0) {
        unsigned short* dst = &sA[buf][arow * LDSA + aseg * 16];
        if (m0 + arow < M) {
            const unsigned short* src = A + (size_t)(m0 + arow) * lda + k0 + aseg * 16;
            async_ld_b128(src, dst);
            async_ld_b128(src + 8, dst + 8);
        } else {
            reinterpret_cast<v4u_t*>(dst)[0] = zero4;
            reinterpret_cast<v4u_t*>(dst)[1] = zero4;
        }
    };
    auto stageBt = [&](int buf, int k0) {           // TRANSB: B is [N,K], contiguous in k
        unsigned short* dst = &sB[buf][arow * LDSB + aseg * 16];
        if (n0 + arow < N) {
            const unsigned short* src = B + (size_t)(n0 + arow) * ldb + k0 + aseg * 16;
            async_ld_b128(src, dst);
            async_ld_b128(src + 8, dst + 8);
        } else {
            reinterpret_cast<v4u_t*>(dst)[0] = zero4;
            reinterpret_cast<v4u_t*>(dst)[1] = zero4;
        }
    };

    v4u_t btmp0 = zero4, btmp1 = zero4;
    auto loadB = [&](int k0) {                      // non-trans: global -> VGPR
        if (n0 + bnb < N) {
            const unsigned short* src = B + (size_t)(k0 + bkc) * ldb + n0 + bnb;
            btmp0 = gload4(src);
            btmp1 = gload4(src + 8);
        } else {
            btmp0 = zero4;
            btmp1 = zero4;
        }
    };
    auto storeB = [&](int buf) {                    // VGPR -> LDS transposed [n][k]
        unsigned short tmp[16];
        reinterpret_cast<v4u_t*>(tmp)[0] = btmp0;
        reinterpret_cast<v4u_t*>(tmp)[1] = btmp1;
#pragma unroll
        for (int i = 0; i < 16; i++) sB[buf][(bnb + i) * LDSB + bkc] = tmp[i];
    };

    const int nkt = (K + BK - 1) / BK;

    // ---- prologue: stage tile 0 into buffer 0
    stageA(0, 0);
    if (TRANSB) {
        stageBt(0, 0);
    } else {
        loadB(0);
        storeB(0);
    }

    for (int kt = 0; kt < nkt; kt++) {
        const int cur = kt & 1;
        const int nxt = cur ^ 1;
        const bool more = (kt + 1) < nkt;

        wait_async0();
        __syncthreads();                            // buf[cur] visible to all waves

        // prefetch next tile while computing this one
        if (more) {
            stageA(nxt, (kt + 1) * BK);
            if (TRANSB) stageBt(nxt, (kt + 1) * BK);
            else        loadB((kt + 1) * BK);
        }

        // ---- fragments per ISA 16-bit A/B layout (preload all, then 8 WMMAs)
        FragBF16 afr[2];
#pragma unroll
        for (int mi = 0; mi < 2; mi++) {
            const int row = wy * 32 + mi * 16 + l16;
#pragma unroll
            for (int i = 0; i < 8; i++) {
                const int kk = (i < 4 ? 0 : 16) + lhalf * 8 + (i & 3) * 2;
                afr[mi].u[i] =
                    *reinterpret_cast<const unsigned int*>(&sA[cur][row * LDSA + kk]);
            }
        }
        FragBF16 bfr[4];
#pragma unroll
        for (int ni = 0; ni < 4; ni++) {
            const int col = wx * 64 + ni * 16 + l16;
#pragma unroll
            for (int i = 0; i < 8; i++) {
                const int kk = (i < 4 ? 0 : 16) + lhalf * 8 + (i & 3) * 2;
                bfr[ni].u[i] =
                    *reinterpret_cast<const unsigned int*>(&sB[cur][col * LDSB + kk]);
            }
        }
#pragma unroll
        for (int ni = 0; ni < 4; ni++)
#pragma unroll
            for (int mi = 0; mi < 2; mi++) {
                acc[mi][ni] = __builtin_amdgcn_wmma_f32_16x16x32_bf16(
                    false, afr[mi].v, false, bfr[ni].v, (short)0, acc[mi][ni], false, false);
            }

        if (more && !TRANSB) storeB(nxt);           // finish staging next B tile
    }

    // ---- epilogue: C layout VGPR j -> M = j + 8*(lane>=16), N = lane%16
    const bool outBf16 = (flags & 1) != 0;
    const bool doElu   = (flags & 2) != 0;
    float*          Cf = reinterpret_cast<float*>(C) + blockIdx.z * cBatch;
    unsigned short* Cb = reinterpret_cast<unsigned short*>(C) + blockIdx.z * cBatch;
#pragma unroll
    for (int mi = 0; mi < 2; mi++)
#pragma unroll
        for (int ni = 0; ni < 4; ni++)
#pragma unroll
            for (int j = 0; j < 8; j++) {
                const int gm = m0 + wy * 32 + mi * 16 + lhalf * 8 + j;
                const int gn = n0 + wx * 64 + ni * 16 + l16;
                if (gm < M && gn < N) {
                    float v = acc[mi][ni][j] * alpha;
                    if (bias) v += bias[gn];
                    if (doElu) v = v > 0.0f ? v : (expf(v) - 1.0f);
                    const size_t idx = (size_t)gm * ldc + (size_t)gn * cstride;
                    if (outBf16) Cb[idx] = f2bf(v);
                    else         Cf[idx] = v;
                }
            }
}

// ---------------------------------------------------------------------------
// Row softmax (fp32 in, bf16 out). One block per row.
// ---------------------------------------------------------------------------
__global__ __launch_bounds__(256)
void softmax_rows(const float* __restrict__ S, unsigned short* __restrict__ P, int cols) {
    __shared__ float red[256];
    const int row = blockIdx.x, tid = threadIdx.x;
    const float* src = S + (size_t)row * cols;
    unsigned short* dst = P + (size_t)row * cols;

    float lmax = -3.4e38f;
    for (int c = tid; c < cols; c += 256) lmax = fmaxf(lmax, src[c]);
    red[tid] = lmax; __syncthreads();
    for (int s = 128; s > 0; s >>= 1) {
        if (tid < s) red[tid] = fmaxf(red[tid], red[tid + s]);
        __syncthreads();
    }
    const float m = red[0]; __syncthreads();

    float lsum = 0.0f;
    for (int c = tid; c < cols; c += 256) lsum += expf(src[c] - m);
    red[tid] = lsum; __syncthreads();
    for (int s = 128; s > 0; s >>= 1) {
        if (tid < s) red[tid] += red[tid + s];
        __syncthreads();
    }
    const float inv = 1.0f / red[0];
    for (int c = tid; c < cols; c += 256) dst[c] = f2bf(expf(src[c] - m) * inv);
}

// ---------------------------------------------------------------------------
// out = LayerNorm(a + b) * g + be ; fp32 out, optional bf16 mirror.
// One block per row of width D.
// ---------------------------------------------------------------------------
__global__ __launch_bounds__(256)
void add_layernorm(const float* __restrict__ a, const float* __restrict__ b,
                   const float* __restrict__ g, const float* __restrict__ be,
                   float* __restrict__ outf, unsigned short* __restrict__ outb, int D) {
    __shared__ float red[256];
    const int row = blockIdx.x, tid = threadIdx.x;
    const float* pa = a + (size_t)row * D;
    const float* pb = b + (size_t)row * D;

    float s = 0.0f, s2 = 0.0f;
    for (int c = tid; c < D; c += 256) {
        const float v = pa[c] + pb[c];
        s += v; s2 += v * v;
    }
    red[tid] = s; __syncthreads();
    for (int st = 128; st > 0; st >>= 1) { if (tid < st) red[tid] += red[tid + st]; __syncthreads(); }
    const float mean = red[0] / D; __syncthreads();
    red[tid] = s2; __syncthreads();
    for (int st = 128; st > 0; st >>= 1) { if (tid < st) red[tid] += red[tid + st]; __syncthreads(); }
    const float var  = red[0] / D - mean * mean;
    const float rstd = rsqrtf(var + 1e-5f);

    for (int c = tid; c < D; c += 256) {
        const float v = pa[c] + pb[c];
        const float y = (v - mean) * rstd * g[c] + be[c];
        outf[(size_t)row * D + c] = y;
        if (outb) outb[(size_t)row * D + c] = f2bf(y);
    }
}

// ---------------------------------------------------------------------------
// Host-side orchestration
// ---------------------------------------------------------------------------
extern "C" void kernel_launch(void* const* d_in, const int* in_sizes, int n_in,
                              void* d_out, int out_size, void* d_ws, size_t ws_size,
                              hipStream_t stream) {
    constexpr int Bb = 4, Ss = 1024, Dd = 512, Ee = 64, Hh = 8;
    constexpr int BS = Bb * Ss;              // 4096 rows
    constexpr int DH = Dd * Hh;              // 4096 interleaved cols

    const float* x   = (const float*)d_in[0];
    const float* Wq  = (const float*)d_in[1];
    const float* Wk  = (const float*)d_in[2];
    const float* Wv  = (const float*)d_in[3];
    const float* Wo  = (const float*)d_in[4];
    const float* W1  = (const float*)d_in[5];
    const float* b1  = (const float*)d_in[6];
    const float* W2  = (const float*)d_in[7];
    const float* b2  = (const float*)d_in[8];
    const float* g1  = (const float*)d_in[9];
    const float* be1 = (const float*)d_in[10];
    const float* g2  = (const float*)d_in[11];
    const float* be2 = (const float*)d_in[12];
    float* out = (float*)d_out;

    // ---- carve workspace (256B aligned blocks)
    char* p = (char*)d_ws;
    auto carve = [&](size_t bytes) -> void* {
        void* r = (void*)p;
        p += (bytes + 255) & ~(size_t)255;
        return r;
    };
    unsigned short* xb   = (unsigned short*)carve((size_t)BS * Dd * 2);
    unsigned short* Wqb  = (unsigned short*)carve((size_t)Hh * Dd * Ee * 2);
    unsigned short* Wkb  = (unsigned short*)carve((size_t)Hh * Dd * Ee * 2);
    unsigned short* Wvb  = (unsigned short*)carve((size_t)Hh * Dd * Dd * 2);
    unsigned short* Wob  = (unsigned short*)carve((size_t)DH * Dd * 2);
    unsigned short* W1b  = (unsigned short*)carve((size_t)Dd * Dd * 2);
    unsigned short* W2b  = (unsigned short*)carve((size_t)Dd * Dd * 2);
    unsigned short* Qh   = (unsigned short*)carve((size_t)Hh * BS * Ee * 2);
    unsigned short* Kh   = (unsigned short*)carve((size_t)Hh * BS * Ee * 2);
    unsigned short* Vh   = (unsigned short*)carve((size_t)Hh * BS * Dd * 2);
    float*          sc   = (float*)         carve((size_t)Bb * Ss * Ss * 4);
    unsigned short* Pm   = (unsigned short*)carve((size_t)Bb * Ss * Ss * 2);
    unsigned short* aO   = (unsigned short*)carve((size_t)BS * DH * 2);
    float*          attn = (float*)         carve((size_t)BS * Dd * 4);
    float*          x1   = (float*)         carve((size_t)BS * Dd * 4);
    unsigned short* x1b  = (unsigned short*)carve((size_t)BS * Dd * 2);
    unsigned short* hb   = (unsigned short*)carve((size_t)BS * Dd * 2);
    float*          ffn  = (float*)         carve((size_t)BS * Dd * 4);

    // ---- fp32 -> bf16 conversions
    auto cvt = [&](const float* src, unsigned short* dst, long long n) {
        int blocks = (int)((n + 1023) / 1024);
        if (blocks > 4096) blocks = 4096;
        if (blocks < 1) blocks = 1;
        cvt_f32_bf16<<<blocks, 256, 0, stream>>>(src, dst, n);
    };
    cvt(x,  xb,  (long long)BS * Dd);
    cvt(Wq, Wqb, (long long)Hh * Dd * Ee);
    cvt(Wk, Wkb, (long long)Hh * Dd * Ee);
    cvt(Wv, Wvb, (long long)Hh * Dd * Dd);
    cvt(Wo, Wob, (long long)DH * Dd);
    cvt(W1, W1b, (long long)Dd * Dd);
    cvt(W2, W2b, (long long)Dd * Dd);

    const int OUT_BF16 = 1, DO_ELU = 2;
    auto grid = [](int M, int N, int z) { return dim3((N + BN - 1) / BN, (M + BM - 1) / BM, z); };

    // ---- Q/K/V projections, batched over heads (z = H)
    gemm_bf16_wmma<false><<<grid(BS, Ee, Hh), 256, 0, stream>>>(
        xb, 0LL, Dd, Wqb, (long long)Dd * Ee, Ee,
        (void*)Qh, (long long)BS * Ee, Ee, 1, BS, Ee, Dd, 1.0f, nullptr, OUT_BF16);
    gemm_bf16_wmma<false><<<grid(BS, Ee, Hh), 256, 0, stream>>>(
        xb, 0LL, Dd, Wkb, (long long)Dd * Ee, Ee,
        (void*)Kh, (long long)BS * Ee, Ee, 1, BS, Ee, Dd, 1.0f, nullptr, OUT_BF16);
    gemm_bf16_wmma<false><<<grid(BS, Dd, Hh), 256, 0, stream>>>(
        xb, 0LL, Dd, Wvb, (long long)Dd * Dd, Dd,
        (void*)Vh, (long long)BS * Dd, Dd, 1, BS, Dd, Dd, 1.0f, nullptr, OUT_BF16);

    // ---- attention per head; O written directly interleaved (col = d*H + h)
    const float scale = 0.125f;  // 1/sqrt(E=64)
    for (int h = 0; h < Hh; h++) {
        const unsigned short* Qp = Qh + (size_t)h * BS * Ee;
        const unsigned short* Kp = Kh + (size_t)h * BS * Ee;
        const unsigned short* Vp = Vh + (size_t)h * BS * Dd;
        // scores[b] = scale * Q[b] @ K[b]^T   (z = B)
        gemm_bf16_wmma<true><<<grid(Ss, Ss, Bb), 256, 0, stream>>>(
            Qp, (long long)Ss * Ee, Ee, Kp, (long long)Ss * Ee, Ee,
            (void*)sc, (long long)Ss * Ss, Ss, 1, Ss, Ss, Ee, scale, nullptr, 0);
        softmax_rows<<<Bb * Ss, 256, 0, stream>>>(sc, Pm, Ss);
        // O[b] = P[b] @ V[b]  -> aO[:, d*H + h]
        gemm_bf16_wmma<false><<<grid(Ss, Dd, Bb), 256, 0, stream>>>(
            Pm, (long long)Ss * Ss, Ss, Vp, (long long)Ss * Dd, Dd,
            (void*)(aO + h), (long long)Ss * DH, DH, Hh, Ss, Dd, Ss, 1.0f, nullptr, OUT_BF16);
    }

    // ---- output projection over interleaved heads
    gemm_bf16_wmma<false><<<grid(BS, Dd, 1), 256, 0, stream>>>(
        aO, 0LL, DH, Wob, 0LL, Dd,
        (void*)attn, 0LL, Dd, 1, BS, Dd, DH, 1.0f, nullptr, 0);

    // ---- x1 = LN(x + attn)
    add_layernorm<<<BS, 256, 0, stream>>>(x, attn, g1, be1, x1, x1b, Dd);

    // ---- FFN: hb = elu(x1 @ W1 + b1); ffn = hb @ W2 + b2
    gemm_bf16_wmma<false><<<grid(BS, Dd, 1), 256, 0, stream>>>(
        x1b, 0LL, Dd, W1b, 0LL, Dd,
        (void*)hb, 0LL, Dd, 1, BS, Dd, Dd, 1.0f, b1, OUT_BF16 | DO_ELU);
    gemm_bf16_wmma<false><<<grid(BS, Dd, 1), 256, 0, stream>>>(
        hb, 0LL, Dd, W2b, 0LL, Dd,
        (void*)ffn, 0LL, Dd, 1, BS, Dd, Dd, 1.0f, b2, 0);

    // ---- out = LN(x1 + ffn)
    add_layernorm<<<BS, 256, 0, stream>>>(x1, ffn, g2, be2, out, nullptr, Dd);
}
